// SmoothAdaptiveSemanticsEmbedding_86973087744006
// MI455X (gfx1250) — compile-verified, matrix operand order checked by reading, not run on hardware
//
#include <hip/hip_runtime.h>

// ---------------------------------------------------------------------------
// CDNA5 (gfx1250) implementation of SmoothAdaptiveSemanticsEmbedding.
//   ns = nt = 4096, d = 256, c = 65 (padded to 96), wave32, WMMA bf16 16x16x32.
// Pipeline:
//   1. split f32 -> (bf16 hi, bf16 lo) copies of all operands (+pad preds)
//   2. row norms, target column sums
//   3. dist = sqrt(max(|a|^2+|b|^2-2*a.b,0)) via 3-term split-bf16 WMMA GEMM
//   4. sem  = same kernel, K=96 padded
//   5. per-row bitonic sort in LDS, ratio scan -> k, beta, bf16 mask
//   6. pos_sum = mask @ target via 2-term split-bf16 WMMA GEMM, epilogue
//      computes pos_mean and neg_mean = (colsum - pos_sum)/(nt-k-1)
// NOTE: keep the GEMM K-loops branch-free — the gfx1250 scheduler pipelines
// the fragment loads 2 K-steps deep on its own.
// ---------------------------------------------------------------------------

typedef __attribute__((ext_vector_type(16))) __bf16 v16bf;
typedef __attribute__((ext_vector_type(8)))  float  v8f;
typedef __attribute__((ext_vector_type(4)))  float  vf4;

union FragU { v16bf v; vf4 q[2]; };

// Load one 16x32 bf16 WMMA operand fragment (A-style; B uses the same pattern
// with row = column index) from a row-major [rows x ld] bf16 matrix.
// ISA 7.12.2: lanes 0-15 hold K = k0..k0+7 (v0-3) and k0+16..k0+23 (v4-7);
// lanes 16-31 hold K = k0+8.. and k0+24.. -> two contiguous 16B loads.
__device__ inline v16bf ld_frag(const __bf16* base, int ld, int row, int k0, int laneHi) {
    FragU u;
    const __bf16* p = base + (size_t)row * ld + k0 + laneHi * 8;
    u.q[0] = *(const vf4*)(p);
    u.q[1] = *(const vf4*)(p + 16);
    return u.v;
}

// ---------------------------------------------------------------------------
// Prep kernels
// ---------------------------------------------------------------------------

__global__ void split_pad_kernel(const float* __restrict__ src,
                                 __bf16* __restrict__ hi, __bf16* __restrict__ lo,
                                 int rows, int cols, int ldo) {
    int idx = blockIdx.x * blockDim.x + threadIdx.x;
    if (idx >= rows * ldo) return;
    int r = idx / ldo, c = idx - r * ldo;
    float v = (c < cols) ? src[(size_t)r * cols + c] : 0.0f;
    __bf16 h = (__bf16)v;                 // RNE to bf16
    hi[idx] = h;
    lo[idx] = (__bf16)(v - (float)h);     // residual
}

// Transposed split: out[cols][rows] (K-contiguous layout for the mask GEMM B).
__global__ void tsplit_kernel(const float* __restrict__ src,
                              __bf16* __restrict__ hi, __bf16* __restrict__ lo,
                              int rows, int cols) {
    int idx = blockIdx.x * blockDim.x + threadIdx.x;
    if (idx >= rows * cols) return;
    int n = idx / rows, k = idx - n * rows;
    float v = src[(size_t)k * cols + n];
    __bf16 h = (__bf16)v;
    hi[idx] = h;
    lo[idx] = (__bf16)(v - (float)h);
}

__global__ void rownorm_kernel(const float* __restrict__ src, float* __restrict__ out,
                               int rows, int cols) {
    int r = blockIdx.x * blockDim.x + threadIdx.x;
    if (r >= rows) return;
    float s = 0.0f;
    for (int c = 0; c < cols; ++c) { float v = src[(size_t)r * cols + c]; s += v * v; }
    out[r] = s;
}

__global__ void colsum_kernel(const float* __restrict__ src, float* __restrict__ out,
                              int rows, int cols) {
    int n = blockIdx.x * blockDim.x + threadIdx.x;
    if (n >= cols) return;
    float s = 0.0f;
    for (int k = 0; k < rows; ++k) s += src[(size_t)k * cols + n];
    out[n] = s;
}

// ---------------------------------------------------------------------------
// Split-bf16 WMMA Euclidean-distance GEMM:
//   out[m][n] = sqrt(max(a2[m] + b2[n] - 2 * dot(A[m], B[n]), 0))
// A: [M x K] row-major (hi/lo), B: [N x K] row-major (hi/lo).
// WG = 256 threads = 8 waves; WG tile 64(M) x 128(N); wave tile 16 x 64.
// ---------------------------------------------------------------------------
__global__ __launch_bounds__(256) void wmma_dist_kernel(
    const __bf16* __restrict__ Ahi, const __bf16* __restrict__ Alo,
    const __bf16* __restrict__ Bhi, const __bf16* __restrict__ Blo,
    const float* __restrict__ a2, const float* __restrict__ b2,
    float* __restrict__ out, int N, int K) {
    const int lane = threadIdx.x & 31;
    const int wave = threadIdx.x >> 5;
    const int wm = wave & 3, wn = wave >> 2;
    const int m0 = blockIdx.y * 64 + wm * 16;
    const int n0 = blockIdx.x * 128 + wn * 64;
    const int lr = lane & 15, lh = lane >> 4;

    v8f acc[4] = {};
    for (int k0 = 0; k0 < K; k0 += 32) {
        v16bf ah = ld_frag(Ahi, K, m0 + lr, k0, lh);
        v16bf al = ld_frag(Alo, K, m0 + lr, k0, lh);
#pragma unroll
        for (int s = 0; s < 4; ++s) {
            v16bf bh = ld_frag(Bhi, K, n0 + s * 16 + lr, k0, lh);
            v16bf bl = ld_frag(Blo, K, n0 + s * 16 + lr, k0, lh);
            acc[s] = __builtin_amdgcn_wmma_f32_16x16x32_bf16(false, ah, false, bh,
                                                             (short)0, acc[s], false, false);
            acc[s] = __builtin_amdgcn_wmma_f32_16x16x32_bf16(false, ah, false, bl,
                                                             (short)0, acc[s], false, false);
            acc[s] = __builtin_amdgcn_wmma_f32_16x16x32_bf16(false, al, false, bh,
                                                             (short)0, acc[s], false, false);
        }
    }
#pragma unroll
    for (int s = 0; s < 4; ++s) {
        int n = n0 + s * 16 + lr;
        float bn = b2[n];
#pragma unroll
        for (int r = 0; r < 8; ++r) {
            int m = m0 + r + lh * 8;
            float d2 = a2[m] + bn - 2.0f * acc[s][r];
            out[(size_t)m * N + n] = sqrtf(fmaxf(d2, 0.0f));
        }
    }
}

// ---------------------------------------------------------------------------
// Per-row bitonic sort + adaptive-k selection.
// One 512-thread workgroup per row; 4096 u64 keys in LDS (32 KB).
// key = (f32 dist bits << 32) | target index  (dist >= 0 so uint order works,
// low index bits give the stable-argsort tie-break).
// ---------------------------------------------------------------------------
__global__ __launch_bounds__(512) void sort_select_kernel(
    const float* __restrict__ dist, const float* __restrict__ sem,
    __bf16* __restrict__ mask, float* __restrict__ kplus1,
    float* __restrict__ beta_out, int nt) {
    __shared__ unsigned long long keys[4096];
    __shared__ float red[512];
    __shared__ int kmin_s;

    const int row = blockIdx.x;
    const int tid = threadIdx.x;
    const float* drow = dist + (size_t)row * nt;
    const float* srow = sem + (size_t)row * nt;

    float sd = 0.0f, sl = 0.0f;
    for (int j = tid; j < nt; j += 512) {
        float dv = drow[j];
        sd += dv;
        sl += srow[j];
        keys[j] = ((unsigned long long)__float_as_uint(dv) << 32) | (unsigned int)j;
    }
    // deterministic tree reductions for the two row means
    red[tid] = sd; __syncthreads();
    for (int s = 256; s > 0; s >>= 1) { if (tid < s) red[tid] += red[tid + s]; __syncthreads(); }
    const float mu_d = red[0] / (float)nt;
    __syncthreads();
    red[tid] = sl; __syncthreads();
    for (int s = 256; s > 0; s >>= 1) { if (tid < s) red[tid] += red[tid + s]; __syncthreads(); }
    const float mu_l = red[0] / (float)nt;

    // bitonic sort ascending
    for (int size = 2; size <= nt; size <<= 1) {
        for (int stride = size >> 1; stride > 0; stride >>= 1) {
            __syncthreads();
            for (int t = tid; t < (nt >> 1); t += 512) {
                int lo = ((t & ~(stride - 1)) << 1) | (t & (stride - 1));
                int hi2 = lo + stride;
                bool asc = ((lo & size) == 0);
                unsigned long long a = keys[lo], b = keys[hi2];
                if ((a > b) == asc) { keys[lo] = b; keys[hi2] = a; }
            }
        }
    }
    __syncthreads();
    if (tid == 0) kmin_s = nt - 1;
    __syncthreads();

    // first sorted position with ratio > 0
    for (int p = tid; p < nt; p += 512) {
        unsigned long long key = keys[p];
        int j = (int)(key & 0xffffffffull);
        float dv = __uint_as_float((unsigned int)(key >> 32));
        float lv = srow[j];
        float ratio = (dv - mu_d) / (lv - mu_l + 1e-4f);
        if (ratio > 0.0f) atomicMin(&kmin_s, p);
    }
    __syncthreads();
    const int k = kmin_s;

    if (tid == 0) {
        unsigned long long key = keys[k];
        int j = (int)(key & 0xffffffffull);
        float dv = __uint_as_float((unsigned int)(key >> 32));
        float lv = srow[j];
        beta_out[row] = (dv - mu_d) / (lv - mu_l + 1e-4f);
        kplus1[row] = (float)(k + 1);
    }
    // scatter 0/1 mask back to unsorted target order (exact in bf16)
    for (int p = tid; p < nt; p += 512) {
        int j = (int)(keys[p] & 0xffffffffull);
        mask[(size_t)row * nt + j] = (__bf16)(p <= k ? 1.0f : 0.0f);
    }
}

// ---------------------------------------------------------------------------
// pos/neg mean GEMM: pos_sum = Mask @ target via WMMA (mask exact bf16,
// target split hi+lo). Ttr is target transposed [dcols x nt] so B fragments
// are K-contiguous. Epilogue applies both normalizations.
// ---------------------------------------------------------------------------
__global__ __launch_bounds__(256) void wmma_posneg_kernel(
    const __bf16* __restrict__ Mask,
    const __bf16* __restrict__ Thi, const __bf16* __restrict__ Tlo,
    const float* __restrict__ colsum, const float* __restrict__ kplus1,
    float* __restrict__ pos_out, float* __restrict__ neg_out,
    int nt, int dcols) {
    const int lane = threadIdx.x & 31;
    const int wave = threadIdx.x >> 5;
    const int wm = wave & 3, wn = wave >> 2;
    const int m0 = blockIdx.y * 64 + wm * 16;
    const int n0 = blockIdx.x * 128 + wn * 64;
    const int lr = lane & 15, lh = lane >> 4;

    v8f acc[4] = {};
    for (int k0 = 0; k0 < nt; k0 += 32) {
        v16bf a = ld_frag(Mask, nt, m0 + lr, k0, lh);
#pragma unroll
        for (int s = 0; s < 4; ++s) {
            v16bf bh = ld_frag(Thi, nt, n0 + s * 16 + lr, k0, lh);
            v16bf bl = ld_frag(Tlo, nt, n0 + s * 16 + lr, k0, lh);
            acc[s] = __builtin_amdgcn_wmma_f32_16x16x32_bf16(false, a, false, bh,
                                                             (short)0, acc[s], false, false);
            acc[s] = __builtin_amdgcn_wmma_f32_16x16x32_bf16(false, a, false, bl,
                                                             (short)0, acc[s], false, false);
        }
    }
#pragma unroll
    for (int s = 0; s < 4; ++s) {
        int n = n0 + s * 16 + lr;
        float cs = colsum[n];
#pragma unroll
        for (int r = 0; r < 8; ++r) {
            int m = m0 + r + lh * 8;
            float ps = acc[s][r];
            float kf = kplus1[m];
            pos_out[(size_t)m * dcols + n] = ps / kf;
            neg_out[(size_t)m * dcols + n] = (cs - ps) / ((float)nt - kf);
        }
    }
}

// ---------------------------------------------------------------------------
// Host launch
// ---------------------------------------------------------------------------
static inline size_t align_up(size_t x, size_t a) { return (x + a - 1) & ~(a - 1); }

extern "C" void kernel_launch(void* const* d_in, const int* in_sizes, int n_in,
                              void* d_out, int out_size, void* d_ws, size_t ws_size,
                              hipStream_t stream) {
    (void)in_sizes; (void)n_in; (void)out_size; (void)ws_size;
    const float* source = (const float*)d_in[0];   // [4096, 256]
    const float* target = (const float*)d_in[1];   // [4096, 256]
    const float* spred  = (const float*)d_in[2];   // [4096, 65]
    const float* tpred  = (const float*)d_in[3];   // [4096, 65]

    const int ns = 4096, nt = 4096, dcol = 256, cc = 65, cpad = 96;

    // workspace carve-out
    char* ws = (char*)d_ws;
    size_t off = 0;
    auto alloc = [&](size_t bytes) -> void* {
        void* p = ws + off;
        off = align_up(off + bytes, 256);
        return p;
    };
    float*  dist   = (float*) alloc((size_t)ns * nt * 4);        // 64 MB
    float*  sem    = (float*) alloc((size_t)ns * nt * 4);        // 64 MB
    __bf16* Maskb  = (__bf16*)alloc((size_t)ns * nt * 2);        // 32 MB
    __bf16* Xhi    = (__bf16*)alloc((size_t)ns * dcol * 2);
    __bf16* Xlo    = (__bf16*)alloc((size_t)ns * dcol * 2);
    __bf16* Yhi    = (__bf16*)alloc((size_t)nt * dcol * 2);
    __bf16* Ylo    = (__bf16*)alloc((size_t)nt * dcol * 2);
    __bf16* Phi    = (__bf16*)alloc((size_t)ns * cpad * 2);
    __bf16* Plo    = (__bf16*)alloc((size_t)ns * cpad * 2);
    __bf16* Qhi    = (__bf16*)alloc((size_t)nt * cpad * 2);
    __bf16* Qlo    = (__bf16*)alloc((size_t)nt * cpad * 2);
    __bf16* Ttrhi  = (__bf16*)alloc((size_t)dcol * nt * 2);
    __bf16* Ttrlo  = (__bf16*)alloc((size_t)dcol * nt * 2);
    float*  a2     = (float*) alloc(ns * 4);
    float*  b2     = (float*) alloc(nt * 4);
    float*  p2     = (float*) alloc(ns * 4);
    float*  q2     = (float*) alloc(nt * 4);
    float*  csum   = (float*) alloc(dcol * 4);
    float*  kplus1 = (float*) alloc(ns * 4);

    float* pos_out  = (float*)d_out;
    float* neg_out  = pos_out + (size_t)ns * dcol;
    float* beta_out = neg_out + (size_t)ns * dcol;

    // 1. split / pad / transpose
    {
        int n = ns * dcol;
        split_pad_kernel<<<(n + 255) / 256, 256, 0, stream>>>(source, Xhi, Xlo, ns, dcol, dcol);
        split_pad_kernel<<<(n + 255) / 256, 256, 0, stream>>>(target, Yhi, Ylo, nt, dcol, dcol);
        int np = ns * cpad;
        split_pad_kernel<<<(np + 255) / 256, 256, 0, stream>>>(spred, Phi, Plo, ns, cc, cpad);
        split_pad_kernel<<<(np + 255) / 256, 256, 0, stream>>>(tpred, Qhi, Qlo, nt, cc, cpad);
        tsplit_kernel<<<(n + 255) / 256, 256, 0, stream>>>(target, Ttrhi, Ttrlo, nt, dcol);
    }
    // 2. norms + column sums
    rownorm_kernel<<<(ns + 255) / 256, 256, 0, stream>>>(source, a2, ns, dcol);
    rownorm_kernel<<<(nt + 255) / 256, 256, 0, stream>>>(target, b2, nt, dcol);
    rownorm_kernel<<<(ns + 255) / 256, 256, 0, stream>>>(spred, p2, ns, cc);
    rownorm_kernel<<<(nt + 255) / 256, 256, 0, stream>>>(tpred, q2, nt, cc);
    colsum_kernel<<<1, 256, 0, stream>>>(target, csum, nt, dcol);

    // 3/4. WMMA distance GEMMs
    {
        dim3 grid(nt / 128, ns / 64);
        wmma_dist_kernel<<<grid, 256, 0, stream>>>(Xhi, Xlo, Yhi, Ylo, a2, b2, dist, nt, dcol);
        wmma_dist_kernel<<<grid, 256, 0, stream>>>(Phi, Plo, Qhi, Qlo, p2, q2, sem, nt, cpad);
    }
    // 5. per-row sort + selection
    sort_select_kernel<<<ns, 512, 0, stream>>>(dist, sem, Maskb, kplus1, beta_out, nt);

    // 6. pos/neg means via WMMA mask GEMM
    {
        dim3 grid(dcol / 128, ns / 64);
        wmma_posneg_kernel<<<grid, 256, 0, stream>>>(Maskb, Ttrhi, Ttrlo, csum, kplus1,
                                                     pos_out, neg_out, nt, dcol);
    }
}
